// LocalGlobalVideoTextInteractions_32229434589945
// MI455X (gfx1250) — compile-verified, matrix operand order checked by reading, not run on hardware
//
#include <hip/hip_runtime.h>
#include <hip/hip_bf16.h>
#include <math.h>

typedef __bf16 bf16;
typedef __attribute__((ext_vector_type(16))) __bf16 v16bf;
typedef __attribute__((ext_vector_type(8)))  float  v8f;
typedef int v4i __attribute__((vector_size(16)));

#define LL 512
#define DD 512
#define BB 16
#define NN 3
#define HH 8
#define DH 64
#define LD (LL*DD)            // 262144

#define GAS __attribute__((address_space(1)))
#define LAS __attribute__((address_space(3)))

union FragBF { v16bf v; uint32_t u[8]; bf16 e[16]; };

// 16x32 bf16 A-fragment (also used for B via W^T), row-major source, ld in elems.
// ISA layout: lanes0-15 row=lane, K 0-7 & 16-23; lanes16-31 row=lane-16, K 8-15 & 24-31.
__device__ inline v16bf load_fragA(const bf16* __restrict__ base, int ld, int row0, int k0) {
  int lane = threadIdx.x & 31;
  int r    = row0 + (lane & 15);
  int hi4  = (lane >> 4) * 4;
  const uint32_t* p = reinterpret_cast<const uint32_t*>(base + (size_t)r * ld + k0);
  FragBF f;
#pragma unroll
  for (int i = 0; i < 4; i++) f.u[i]     = p[hi4 + i];
#pragma unroll
  for (int i = 0; i < 4; i++) f.u[4 + i] = p[8 + hi4 + i];
  return f.v;
}

__device__ inline v8f wmma_bf16(v16bf a, v16bf b, v8f c) {
  return __builtin_amdgcn_wmma_f32_16x16x32_bf16(false, a, false, b, (short)0, c, false, false);
}

// async global -> LDS copy, 16B per lane (ASYNCcnt-tracked on gfx1250)
__device__ inline void async_g2l_b128(void* lds, const void* g) {
#if __has_builtin(__builtin_amdgcn_global_load_async_to_lds_b128)
  __builtin_amdgcn_global_load_async_to_lds_b128((GAS v4i*)(g), (LAS v4i*)(lds), 0, 0);
#else
  *(uint4*)lds = *(const uint4*)g;
#endif
}
__device__ inline void wait_async_lds() {
#if __has_builtin(__builtin_amdgcn_global_load_async_to_lds_b128)
  asm volatile("s_wait_asynccnt 0" ::: "memory");
#endif
}

__device__ inline float rmax16(float v) {
#pragma unroll
  for (int m = 1; m < 16; m <<= 1) v = fmaxf(v, __shfl_xor(v, m, 32));
  return v;
}
__device__ inline float rsum16(float v) {
#pragma unroll
  for (int m = 1; m < 16; m <<= 1) v += __shfl_xor(v, m, 32);
  return v;
}

// ---------------- weight transpose + bf16 convert:  W(z,K,N) -> WT(z,N,K) ----------------
__global__ void transpose_cvt_kernel(const float* __restrict__ W, bf16* __restrict__ WT,
                                     int K, int N, int total) {
  int i = blockIdx.x * 256 + threadIdx.x;
  if (i >= total) return;
  int kn = K * N;
  int z = i / kn, rem = i - z * kn;
  int k = rem / N, n = rem - k * N;
  WT[(size_t)z * kn + (size_t)n * K + k] = (bf16)W[i];
}

// ---------------- generic f32 -> bf16 (batch-strided source) ----------------
__global__ void cvt_bf16_kernel(const float* __restrict__ src, long srcStride,
                                bf16* __restrict__ dst, int per, int nb) {
  int i = blockIdx.x * 256 + threadIdx.x;
  if (i >= per * nb) return;
  int z = i / per, r = i - z * per;
  dst[i] = (bf16)src[(size_t)z * srcStride + r];
}

// ---------------- h2[b,n,k] = relu(se[b,n,:] @ w2[n,:,k] + b2[n,k]) ----------------
__global__ void h2_kernel(const float* __restrict__ se, const float* __restrict__ w2,
                          const float* __restrict__ b2, float* __restrict__ h2o, int total) {
  int i = blockIdx.x * 256 + threadIdx.x;
  if (i >= total) return;
  int k = i & (DD - 1);
  int zn = i >> 9;           // b*3+n
  int n  = zn % NN;
  const float* s = se + (size_t)zn * DD;
  const float* w = w2 + (size_t)n * DD * DD + k;
  float acc = b2[n * DD + k];
  for (int d = 0; d < DD; d++) acc += s[d] * w[(size_t)d * DD];
  h2o[i] = fmaxf(acc, 0.f);
}

// ---------------- batched WMMA GEMM with async double-buffered LDS staging ----------------
// C[z] = epi(A[z/aDiv] @ WT[z%wMod]^T + bias); block = 256 thr = 8 waves; tile 64x64, K-step 32.
__global__ __launch_bounds__(256) void gemm_bf16_kernel(
    const bf16* __restrict__ A, long aStride, int aDiv,
    const bf16* __restrict__ BT, long wStride, int wMod,
    const float* __restrict__ bias, int biasStride,
    const float* __restrict__ colScale, long csStride,
    float* __restrict__ Cf, bf16* __restrict__ Cb, long cStride,
    int N, int K, int relu) {
  __shared__ bf16 lA[2][64 * 32];
  __shared__ bf16 lB[2][64 * 32];
  int z = blockIdx.z;
  const bf16* Az = A + (size_t)(z / aDiv) * aStride;
  int widx = (wMod > 1) ? (z % wMod) : 0;
  const bf16*  Bw = BT + (size_t)widx * wStride;
  const float* bz = bias + (size_t)widx * biasStride;
  int w  = threadIdx.x >> 5;
  int mi = w & 3, ni = w >> 2;
  int rowB = blockIdx.y * 64;          // block row base
  int colB = blockIdx.x * 64;          // block col base
  // staging slot: each thread moves one 16B chunk of A-tile and one of B-tile
  int tid   = threadIdx.x;
  int srow  = tid >> 2;                // 0..63
  int scol  = (tid & 3) * 8;           // bf16 element offset (8 elems = 16B)
  v8f acc0 = {0.f,0.f,0.f,0.f,0.f,0.f,0.f,0.f};
  v8f acc1 = acc0;

  // prologue: stage K-chunk 0 into buffer 0
  async_g2l_b128(&lA[0][srow * 32 + scol], Az + (size_t)(rowB + srow) * K + scol);
  async_g2l_b128(&lB[0][srow * 32 + scol], Bw + (size_t)(colB + srow) * K + scol);

  int steps = K >> 5;
  for (int it = 0; it < steps; it++) {
    int p = it & 1;
    wait_async_lds();
    __syncthreads();                   // buf p ready; buf p^1 free for restage
    if (it + 1 < steps) {
      int kk = (it + 1) << 5;
      async_g2l_b128(&lA[p ^ 1][srow * 32 + scol], Az + (size_t)(rowB + srow) * K + kk + scol);
      async_g2l_b128(&lB[p ^ 1][srow * 32 + scol], Bw + (size_t)(colB + srow) * K + kk + scol);
    }
    v16bf a  = load_fragA(lA[p], 32, mi * 16, 0);
    v16bf b0 = load_fragA(lB[p], 32, ni * 32, 0);
    v16bf b1 = load_fragA(lB[p], 32, ni * 32 + 16, 0);
    acc0 = wmma_bf16(a, b0, acc0);
    acc1 = wmma_bf16(a, b1, acc1);
  }

  int lane = threadIdx.x & 31;
  int cc = lane & 15;
  int radd = (lane >> 4) * 8;
  int row0 = rowB + mi * 16;
  int col0 = colB + ni * 32;
  const float* cs = colScale ? colScale + (size_t)z * csStride : nullptr;
#pragma unroll
  for (int r = 0; r < 8; r++) {
    int row = row0 + r + radd;
#pragma unroll
    for (int t = 0; t < 2; t++) {
      int col = col0 + t * 16 + cc;
      float v = (t ? acc1[r] : acc0[r]) + bz[col];
      if (relu) v = fmaxf(v, 0.f);
      if (cs)   v *= cs[col];
      size_t o = (size_t)z * cStride + (size_t)row * N + col;
      if (Cf) Cf[o] = v;
      if (Cb) Cb[o] = (bf16)v;
    }
  }
}

// ---------------- local banded attention (15-wide), in-place residual ----------------
__global__ __launch_bounds__(256) void band_attn_kernel(
    float* __restrict__ x, long xStride,
    const float* __restrict__ mk, const float* __restrict__ ct,
    const float* __restrict__ mask) {
  __shared__ float sS[HH][16];
  int b = blockIdx.y, l = blockIdx.x, t = threadIdx.x;
  const float* mkb = mk + (size_t)b * LD + (size_t)l * DD;
  const float* ctb = ct + (size_t)b * LL * 1024;
  if (t < HH * 15) {
    int h = t / 15, jj = t - h * 15, j = l + jj - 7;
    float s = -1e9f;
    if (j >= 0 && j < LL && mask[b * LL + j] != 0.f) {
      const float* q  = ctb + (size_t)j * 1024 + h * DH;
      const float* kx = mkb + h * DH;
      float d = 0.f;
      for (int e = 0; e < DH; e++) d += kx[e] * q[e];
      s = d * 0.125f;
    }
    sS[h][jj] = s;
  }
  __syncthreads();
  if (t < HH) {
    float mx = -1e30f;
    for (int jj = 0; jj < 15; jj++) mx = fmaxf(mx, sS[t][jj]);
    float sm = 0.f;
    for (int jj = 0; jj < 15; jj++) { float e = __expf(sS[t][jj] - mx); sS[t][jj] = e; sm += e; }
    float inv = 1.f / sm;
    for (int jj = 0; jj < 15; jj++) sS[t][jj] *= inv;
  }
  __syncthreads();
#pragma unroll
  for (int u = 0; u < 2; u++) {
    int d = t * 2 + u;
    int h = d >> 6;
    float r = 0.f;
    for (int jj = 0; jj < 15; jj++) {
      int j = l + jj - 7;
      if (j >= 0 && j < LL) r += sS[h][jj] * ctb[(size_t)j * 1024 + 512 + d];
    }
    x[(size_t)b * xStride + (size_t)l * DD + d] += r;
  }
}

// ---------------- satt: alpha[bn] = tanh(se@w1) @ w2 ----------------
__global__ __launch_bounds__(256) void satt_alpha_kernel(
    const float* __restrict__ se, const float* __restrict__ w1,
    const float* __restrict__ w2, float* __restrict__ alpha) {
  __shared__ float red[256];
  int bn = blockIdx.x, j = threadIdx.x;
  const float* s = se + (size_t)bn * DD;
  float acc = 0.f;
  for (int d = 0; d < DD; d++) acc += s[d] * w1[d * 256 + j];
  red[j] = tanhf(acc) * w2[j];
  __syncthreads();
  for (int st = 128; st > 0; st >>= 1) { if (j < st) red[j] += red[j + st]; __syncthreads(); }
  if (j == 0) alpha[bn] = red[0];
}

__global__ void satt_softmax_kernel(const float* __restrict__ alpha,
                                    float* __restrict__ sw, float* __restrict__ tail) {
  int b = threadIdx.x;
  if (b >= BB) return;
  float a0 = alpha[b*3], a1 = alpha[b*3+1], a2 = alpha[b*3+2];
  float m = fmaxf(a0, fmaxf(a1, a2));
  float e0 = __expf(a0-m), e1 = __expf(a1-m), e2 = __expf(a2-m);
  float inv = 1.f / (e0 + e1 + e2);
  sw[b*3] = e0*inv; sw[b*3+1] = e1*inv; sw[b*3+2] = e2*inv;
  tail[b*3] = e0*inv; tail[b*3+1] = e1*inv; tail[b*3+2] = e2*inv;
}

// ---------------- a[b,l,d] = sum_n sw[b,n] * ss[b,n,l,d] ----------------
__global__ void combine_kernel(const float* __restrict__ ss, const float* __restrict__ sw,
                               float* __restrict__ xa) {
  int i = blockIdx.x * 256 + threadIdx.x;
  if (i >= BB * LD) return;
  int b = i / LD, r = i - b * LD;
  const float* base = ss + (size_t)(b * 3) * LD + r;
  xa[i] = sw[b*3] * base[0] + sw[b*3+1] * base[LD] + sw[b*3+2] * base[2*LD];
}

// ---------------- mvT[b,h,d,m] = (bf16) ct[b,m,512+h*64+d] ----------------
__global__ void mvt_kernel(const float* __restrict__ ctf, bf16* __restrict__ mvT) {
  int i = blockIdx.x * 256 + threadIdx.x;
  if (i >= BB * LD) return;
  int m = i & 511, d = (i >> 9) & 63, h = (i >> 15) & 7, b = i >> 18;
  mvT[i] = (bf16)ctf[((size_t)b * LL + m) * 1024 + 512 + h * DH + d];
}

// ---------------- flash attention (global NL): x += softmax(mk@mq^T/8 masked) @ mv ----------------
__global__ __launch_bounds__(256) void flash_attn_kernel(
    const bf16* __restrict__ mkb, const bf16* __restrict__ ctb,
    const bf16* __restrict__ mvT, const float* __restrict__ mask,
    float* __restrict__ x) {
  __shared__ bf16 pl[8][16 * 32];
  int w = threadIdx.x >> 5, lane = threadIdx.x & 31;
  int b = blockIdx.z, h = blockIdx.y;
  int ltile = blockIdx.x * 8 + w;
  int row0 = ltile * 16;
  const bf16* mkh = mkb + (size_t)b * LD + h * DH;              // rows l, ld 512
  const bf16* mqh = ctb + (size_t)b * LL * 1024 + h * DH;       // rows m, ld 1024
  const bf16* mvh = mvT + (size_t)(b * HH + h) * DH * LL;       // rows d, ld 512 (k = m)
  v16bf amk0 = load_fragA(mkh, DD, row0, 0);
  v16bf amk1 = load_fragA(mkh, DD, row0, 32);
  const v8f vzero = {0.f,0.f,0.f,0.f,0.f,0.f,0.f,0.f};
  v8f racc[4] = {vzero, vzero, vzero, vzero};
  float Mr[8], Sr[8];
#pragma unroll
  for (int r = 0; r < 8; r++) { Mr[r] = -1e30f; Sr[r] = 0.f; }
  int cc = lane & 15, rsel = lane >> 4;
  bf16* tile = pl[w];
  for (int m0 = 0; m0 < LL; m0 += 32) {
    v16bf bq0a = load_fragA(mqh, 1024, m0,      0);
    v16bf bq0b = load_fragA(mqh, 1024, m0,      32);
    v16bf bq1a = load_fragA(mqh, 1024, m0 + 16, 0);
    v16bf bq1b = load_fragA(mqh, 1024, m0 + 16, 32);
    v8f s0 = vzero, s1 = vzero;
    s0 = wmma_bf16(amk0, bq0a, s0); s0 = wmma_bf16(amk1, bq0b, s0);
    s1 = wmma_bf16(amk0, bq1a, s1); s1 = wmma_bf16(amk1, bq1b, s1);
    float msk0 = mask[b * LL + m0 + cc];
    float msk1 = mask[b * LL + m0 + 16 + cc];
#pragma unroll
    for (int r = 0; r < 8; r++) {
      float v0 = s0[r] * 0.125f; if (msk0 == 0.f) v0 = -1e9f;
      float v1 = s1[r] * 0.125f; if (msk1 == 0.f) v1 = -1e9f;
      float rm = rmax16(fmaxf(v0, v1));
      float Mn = fmaxf(Mr[r], rm);
      float p0 = __expf(v0 - Mn), p1 = __expf(v1 - Mn);
      float ps = rsum16(p0 + p1);
      float corr = __expf(Mr[r] - Mn);
      Sr[r] = Sr[r] * corr + ps;
      Mr[r] = Mn;
#pragma unroll
      for (int tt = 0; tt < 4; tt++) racc[tt][r] *= corr;
      int rowi = r + rsel * 8;
      tile[rowi * 32 + cc]      = (bf16)p0;
      tile[rowi * 32 + 16 + cc] = (bf16)p1;
    }
    asm volatile("s_wait_dscnt 0" ::: "memory");
    v16bf pA = load_fragA(tile, 32, 0, 0);
#pragma unroll
    for (int tt = 0; tt < 4; tt++)
      racc[tt] = wmma_bf16(pA, load_fragA(mvh, LL, tt * 16, m0), racc[tt]);
  }
#pragma unroll
  for (int r = 0; r < 8; r++) {
    float inv = 1.f / Sr[r];
    int l = row0 + r + rsel * 8;
#pragma unroll
    for (int tt = 0; tt < 4; tt++) {
      int d = h * DH + tt * 16 + cc;
      size_t o = (size_t)b * LD + (size_t)l * DD + d;
      x[o] += racc[tt][r] * inv;
    }
  }
}

__global__ void copy_kernel(const float* __restrict__ src, float* __restrict__ dst, int n) {
  int i = blockIdx.x * 256 + threadIdx.x;
  if (i < n) dst[i] = src[i];
}

// =======================================================================================
extern "C" void kernel_launch(void* const* d_in, const int* in_sizes, int n_in,
                              void* d_out, int out_size, void* d_ws, size_t ws_size,
                              hipStream_t stream) {
  const float* seg   = (const float*)d_in[0];
  const float* smask = (const float*)d_in[1];
  const float* se    = (const float*)d_in[2];
  const float* hpw1  = (const float*)d_in[3];
  const float* hpb1  = (const float*)d_in[4];
  const float* hpw2  = (const float*)d_in[5];
  const float* hpb2  = (const float*)d_in[6];
  const float* hpw3  = (const float*)d_in[7];
  const float* hpb3  = (const float*)d_in[8];
  const float* lcW   = (const float*)d_in[9];
  const float* lcb   = (const float*)d_in[10];
  const float* lvW   = (const float*)d_in[11];
  const float* lvb   = (const float*)d_in[12];
  const float* gcW   = (const float*)d_in[13];
  const float* gcb   = (const float*)d_in[14];
  const float* gvW   = (const float*)d_in[15];
  const float* gvb   = (const float*)d_in[16];
  const float* sw1   = (const float*)d_in[17];
  const float* sw2   = (const float*)d_in[18];
  float* out = (float*)d_out;

  char* p = (char*)d_ws;
  auto alloc = [&](size_t bytes) -> void* {
    void* r = (void*)p;
    p += (bytes + 255) & ~(size_t)255;
    return r;
  };
  bf16*  w1T  = (bf16*)alloc((size_t)3 * 262144 * 2);
  bf16*  w3T  = (bf16*)alloc((size_t)3 * 262144 * 2);
  bf16*  lcWT = (bf16*)alloc((size_t)6 * 524288 * 2);
  bf16*  lvWT = (bf16*)alloc((size_t)6 * 262144 * 2);
  bf16*  gcWT = (bf16*)alloc((size_t)2 * 524288 * 2);
  bf16*  gvWT = (bf16*)alloc((size_t)2 * 262144 * 2);
  float* h2   = (float*)alloc((size_t)48 * 512 * 4);
  bf16*  xbf  = (bf16*)alloc((size_t)BB * LD * 2);
  bf16*  ybf  = (bf16*)alloc((size_t)48 * LD * 2);
  float* ss   = (float*)alloc((size_t)48 * LD * 4);
  float* mkf  = (float*)alloc((size_t)BB * LD * 4);
  bf16*  mkbf = (bf16*)alloc((size_t)BB * LD * 2);
  float* ctf  = (float*)alloc((size_t)BB * LL * 1024 * 4);
  bf16*  ctbf = (bf16*)alloc((size_t)BB * LL * 1024 * 2);
  bf16*  mvTb = (bf16*)alloc((size_t)BB * LD * 2);
  float* xa   = (float*)alloc((size_t)BB * LD * 4);
  float* alph = (float*)alloc(256 * 4);
  float* swb  = (float*)alloc(256 * 4);

  auto blk = [](int tot) { return dim3((tot + 255) / 256); };

  // --- weights: transpose + bf16 ---
  transpose_cvt_kernel<<<blk(3*262144), 256, 0, stream>>>(hpw1, w1T, 512, 512, 3*262144);
  transpose_cvt_kernel<<<blk(3*262144), 256, 0, stream>>>(hpw3, w3T, 512, 512, 3*262144);
  transpose_cvt_kernel<<<blk(6*524288), 256, 0, stream>>>(lcW,  lcWT, 512, 1024, 6*524288);
  transpose_cvt_kernel<<<blk(6*262144), 256, 0, stream>>>(lvW,  lvWT, 512, 512, 6*262144);
  transpose_cvt_kernel<<<blk(2*524288), 256, 0, stream>>>(gcW,  gcWT, 512, 1024, 2*524288);
  transpose_cvt_kernel<<<blk(2*262144), 256, 0, stream>>>(gvW,  gvWT, 512, 512, 2*262144);

  // --- stage A: hyper-projection ---
  cvt_bf16_kernel<<<blk(BB*LD), 256, 0, stream>>>(seg, LD, xbf, LD, BB);
  h2_kernel<<<blk(48*512), 256, 0, stream>>>(se, hpw2, hpb2, h2, 48*512);
  // y = relu(x @ W1 + b1) * h2   (Z = 48 = b*3+n)
  gemm_bf16_kernel<<<dim3(8,8,48), 256, 0, stream>>>(
      xbf, (long)LD, 3, w1T, 262144L, 3, hpb1, 512,
      h2, 512L, nullptr, ybf, (long)LD, 512, 512, 1);
  // ss = relu(y @ W3 + b3)
  gemm_bf16_kernel<<<dim3(8,8,48), 256, 0, stream>>>(
      ybf, (long)LD, 1, w3T, 262144L, 3, hpb3, 512,
      nullptr, 0L, ss, nullptr, (long)LD, 512, 512, 1);

  // --- local NL blocks (banded attention), ss updated in place per n ---
  for (int n = 0; n < NN; n++) {
    for (int s = 0; s < 2; s++) {
      int ws_i = n * 2 + s;
      cvt_bf16_kernel<<<blk(BB*LD), 256, 0, stream>>>(ss + (size_t)n*LD, 3L*LD, xbf, LD, BB);
      gemm_bf16_kernel<<<dim3(8,8,BB), 256, 0, stream>>>(
          xbf, (long)LD, 1, lvWT + (size_t)ws_i*262144, 0L, 1, lvb + ws_i*512, 512,
          nullptr, 0L, mkf, nullptr, (long)LD, 512, 512, 0);
      gemm_bf16_kernel<<<dim3(16,8,BB), 256, 0, stream>>>(
          xbf, (long)LD, 1, lcWT + (size_t)ws_i*524288, 0L, 1, lcb + ws_i*1024, 1024,
          nullptr, 0L, ctf, nullptr, (long)(LL*1024), 1024, 512, 0);
      band_attn_kernel<<<dim3(LL, BB), 256, 0, stream>>>(
          ss + (size_t)n*LD, 3L*LD, mkf, ctf, smask);
    }
  }

  // --- attention pooling over n ---
  satt_alpha_kernel<<<dim3(48), 256, 0, stream>>>(se, sw1, sw2, alph);
  satt_softmax_kernel<<<dim3(1), 32, 0, stream>>>(alph, swb, out + (size_t)BB*LD);
  combine_kernel<<<blk(BB*LD), 256, 0, stream>>>(ss, swb, xa);

  // --- global NL blocks (flash attention), xa updated in place ---
  for (int s = 0; s < 2; s++) {
    cvt_bf16_kernel<<<blk(BB*LD), 256, 0, stream>>>(xa, (long)LD, xbf, LD, BB);
    gemm_bf16_kernel<<<dim3(8,8,BB), 256, 0, stream>>>(
        xbf, (long)LD, 1, gvWT + (size_t)s*262144, 0L, 1, gvb + s*512, 512,
        nullptr, 0L, nullptr, mkbf, (long)LD, 512, 512, 0);
    gemm_bf16_kernel<<<dim3(16,8,BB), 256, 0, stream>>>(
        xbf, (long)LD, 1, gcWT + (size_t)s*524288, 0L, 1, gcb + s*1024, 1024,
        nullptr, 0L, ctf, ctbf, (long)(LL*1024), 1024, 512, 0);
    mvt_kernel<<<blk(BB*LD), 256, 0, stream>>>(ctf, mvTb);
    flash_attn_kernel<<<dim3(4, HH, BB), 256, 0, stream>>>(mkbf, ctbf, mvTb, smask, xa);
  }

  copy_kernel<<<blk(BB*LD), 256, 0, stream>>>(xa, out, BB*LD);
}
